// WorldModelNcpSingleStep_78176994722244
// MI455X (gfx1250) — matrix operand order repaired and checked by reading, not compiled, not run on hardware
//
#include <hip/hip_runtime.h>
#include <hip/hip_bf16.h>
#include <math.h>

// ---------------------------------------------------------------------------
// WorldModel NCP (CfC) forward for MI455X (gfx1250, wave32, WMMA bf16)
// ---------------------------------------------------------------------------
typedef __bf16 bf16_t;
typedef __attribute__((ext_vector_type(16))) __bf16 v16bf;
typedef __attribute__((ext_vector_type(8)))  __bf16 v8bf;
typedef __attribute__((ext_vector_type(8)))  float  v8f;

#define E_    64
#define S_    17
#define SE_   1088
#define B_    32
#define T_    128
#define BT_   (B_*T_)
#define OBSV_ 512

// A-fragment for V_WMMA_F32_16X16X32_BF16 (16x32, row-major source):
// lane l: row = l&15; khalf = l>>4.
// VGPR0..3 hold K = khalf*16 .. khalf*16+7 ; VGPR4..7 hold K = khalf*16+8 .. +15
__device__ __forceinline__ v16bf load_a_frag(const bf16_t* __restrict__ p, int khalf) {
  v8bf lo = *(const v8bf*)(p + khalf * 16);
  v8bf hi = *(const v8bf*)(p + khalf * 16 + 8);
  v16bf a;
#pragma unroll
  for (int i = 0; i < 8; ++i) { a[i] = lo[i]; a[8 + i] = hi[i]; }
  return a;
}

// ---------------------------------------------------------------------------
// Pack a K x N fp32 weight (optionally elementwise-masked) into the B-fragment
// bf16 layout used here (contiguous k, same order as load_a_frag):
//   dst[ (((ct*nkt + kt)*ngates + gate)*32 + lane)*16 + j ]
// holds element (k = kt*32 + (lane>>4)*16 + j , col = ct*16 + (lane&15)).
// One lane's fragment = 16 contiguous bf16 = 32B -> two global_load_b128.
// ---------------------------------------------------------------------------
__global__ void pack_w_kernel(const float* __restrict__ w, const float* __restrict__ mask,
                              bf16_t* __restrict__ dst, int Kdim, int N, int nkt,
                              int ngates, int gate) {
  size_t tid = (size_t)blockIdx.x * blockDim.x + threadIdx.x;
  size_t total = (size_t)Kdim * N;
  if (tid >= total) return;
  int j    = (int)(tid & 15);
  int lane = (int)((tid >> 4) & 31);
  size_t rest = tid >> 9;
  int kt = (int)(rest % nkt);
  int ct = (int)(rest / nkt);
  int k   = kt * 32 + ((lane >> 4) << 4) + j;
  int col = ct * 16 + (lane & 15);
  float v = w[(size_t)k * N + col];
  if (mask) v *= mask[(size_t)k * N + col];
  dst[((((size_t)ct * nkt + kt) * ngates + gate) * 32 + (size_t)lane) * 16 + j] = (bf16_t)v;
}

// ---------------------------------------------------------------------------
// One CfC layer, one timestep: fused 4-gate GEMM (M=32) + CfC combine.
// Wave = one (column-tile ct, row-tile mt). 4 f32 accumulators = ff1,ff2,ta,tb
// for the same 16 output columns (gate-interleaved packed weights).
// A operand gathered from two sources: [seq/prev-layer input | hidden state].
// ---------------------------------------------------------------------------
__global__ void cfc_step_kernel(
    const bf16_t* __restrict__ A0, long sA0,
    const bf16_t* __restrict__ A1, long sA1,
    int K0len, int nkt,
    const bf16_t* __restrict__ Wp,
    const float* __restrict__ b1, const float* __restrict__ b2,
    const float* __restrict__ ba, const float* __restrict__ bb,
    bf16_t* __restrict__ dst0, long sD0,
    bf16_t* __restrict__ dst1, long sD1,
    float* __restrict__ dstF, long sDF) {
  const int lane  = threadIdx.x & 31;
  const int wave  = blockIdx.x * (blockDim.x >> 5) + (threadIdx.x >> 5);
  const int ct    = wave >> 1;        // column tile within dh
  const int mt    = wave & 1;         // row tile (M = 32 -> 2 tiles)
  const int khalf = lane >> 4;
  const int lrow  = lane & 15;
  const int arow  = mt * 16 + lrow;

  v8f acc0 = {}, acc1 = {}, acc2 = {}, acc3 = {};
  const bf16_t* wbase = Wp + (size_t)ct * nkt * 2048 + (size_t)lane * 16;

  for (int kt = 0; kt < nkt; ++kt) {
    const int k0 = kt * 32;
    const bf16_t* ap = (k0 < K0len) ? (A0 + (size_t)arow * sA0 + k0)
                                    : (A1 + (size_t)arow * sA1 + (k0 - K0len));
    v16bf a = load_a_frag(ap, khalf);
    const bf16_t* wp = wbase + (size_t)kt * 2048;
    if (kt + 1 < nkt) __builtin_prefetch(wp + 2048, 0, 3);
    v16bf bf;
    bf   = *(const v16bf*)(wp + 0 * 512);
    acc0 = __builtin_amdgcn_wmma_f32_16x16x32_bf16(false, a, false, bf, (short)0, acc0, false, false);
    bf   = *(const v16bf*)(wp + 1 * 512);
    acc1 = __builtin_amdgcn_wmma_f32_16x16x32_bf16(false, a, false, bf, (short)0, acc1, false, false);
    bf   = *(const v16bf*)(wp + 2 * 512);
    acc2 = __builtin_amdgcn_wmma_f32_16x16x32_bf16(false, a, false, bf, (short)0, acc2, false, false);
    bf   = *(const v16bf*)(wp + 3 * 512);
    acc3 = __builtin_amdgcn_wmma_f32_16x16x32_bf16(false, a, false, bf, (short)0, acc3, false, false);
  }

  const int colB = ct * 16 + lrow;
  const float vb1 = b1[colB], vb2 = b2[colB], vab = ba[colB] + bb[colB];
#pragma unroll
  for (int r = 0; r < 8; ++r) {
    const int row = mt * 16 + khalf * 8 + r;     // C/D layout: M = r + 8*(lane>=16)
    float g1 = acc0[r] + vb1;
    float g2 = acc1[r] + vb2;
    float z  = acc2[r] + acc3[r] + vab;
    float s  = 1.0f / (1.0f + __expf(-z));
    float hn = tanhf(g1) * (1.0f - s) + s * tanhf(g2);
    if (dst0) dst0[(size_t)row * sD0 + colB] = (bf16_t)hn;
    if (dst1) dst1[(size_t)row * sD1 + colB] = (bf16_t)hn;
    if (dstF) dstF[(size_t)row * sDF + colB] = hn;
  }
}

// ---------------------------------------------------------------------------
// Generic head GEMM: C[M,N] = A[M,K]@Wp + bias, optional ReLU; wave tile 16x64.
// ---------------------------------------------------------------------------
__global__ void head_gemm_kernel(
    const bf16_t* __restrict__ A, long sA, int nkt,
    const bf16_t* __restrict__ Wp, int nGroups,     // nGroups = N/64
    const float* __restrict__ bias,
    float* __restrict__ outF, bf16_t* __restrict__ outB, long sO, int relu) {
  const int lane  = threadIdx.x & 31;
  const int wave  = blockIdx.x * (blockDim.x >> 5) + (threadIdx.x >> 5);
  const int mt    = wave / nGroups;
  const int ng    = wave % nGroups;
  const int khalf = lane >> 4;
  const int lrow  = lane & 15;
  const int arow  = mt * 16 + lrow;

  v8f acc0 = {}, acc1 = {}, acc2 = {}, acc3 = {};
  const bf16_t* wb0 = Wp + ((size_t)(ng * 4 + 0) * nkt) * 512 + (size_t)lane * 16;
  const bf16_t* wb1 = Wp + ((size_t)(ng * 4 + 1) * nkt) * 512 + (size_t)lane * 16;
  const bf16_t* wb2 = Wp + ((size_t)(ng * 4 + 2) * nkt) * 512 + (size_t)lane * 16;
  const bf16_t* wb3 = Wp + ((size_t)(ng * 4 + 3) * nkt) * 512 + (size_t)lane * 16;

  for (int kt = 0; kt < nkt; ++kt) {
    v16bf a = load_a_frag(A + (size_t)arow * sA + kt * 32, khalf);
    const size_t ko = (size_t)kt * 512;
    v16bf bf;
    bf   = *(const v16bf*)(wb0 + ko);
    acc0 = __builtin_amdgcn_wmma_f32_16x16x32_bf16(false, a, false, bf, (short)0, acc0, false, false);
    bf   = *(const v16bf*)(wb1 + ko);
    acc1 = __builtin_amdgcn_wmma_f32_16x16x32_bf16(false, a, false, bf, (short)0, acc1, false, false);
    bf   = *(const v16bf*)(wb2 + ko);
    acc2 = __builtin_amdgcn_wmma_f32_16x16x32_bf16(false, a, false, bf, (short)0, acc2, false, false);
    bf   = *(const v16bf*)(wb3 + ko);
    acc3 = __builtin_amdgcn_wmma_f32_16x16x32_bf16(false, a, false, bf, (short)0, acc3, false, false);
  }

#pragma unroll
  for (int j = 0; j < 4; ++j) {
    v8f acc = (j == 0) ? acc0 : (j == 1) ? acc1 : (j == 2) ? acc2 : acc3;
    const int col = (ng * 4 + j) * 16 + lrow;
    const float b = bias[col];
#pragma unroll
    for (int r = 0; r < 8; ++r) {
      const int row = mt * 16 + khalf * 8 + r;
      float v = acc[r] + b;
      if (relu) v = v > 0.0f ? v : 0.0f;
      if (outF) outF[(size_t)row * sO + col] = v;
      if (outB) outB[(size_t)row * sO + col] = (bf16_t)v;
    }
  }
}

// Tiny second matmuls of head_rew / head_end: (BT x 64) @ (64 x Ncol) + b
__global__ void head_small_kernel(const float* __restrict__ hid, const float* __restrict__ w2,
                                  const float* __restrict__ b2, float* __restrict__ out, int Ncol) {
  size_t tid = (size_t)blockIdx.x * blockDim.x + threadIdx.x;
  if (tid >= (size_t)BT_ * Ncol) return;
  const size_t row = tid / Ncol;
  const int c = (int)(tid % Ncol);
  float s = b2[c];
  const float* h = hid + row * 64;
#pragma unroll 8
  for (int k = 0; k < 64; ++k) s += h[k] * w2[(size_t)k * Ncol + c];
  out[tid] = s;
}

// LayerNorm over SE=1088 per (b,t) row; emits bf16 for WMMA consumption.
__global__ void layernorm_bf16_kernel(const float* __restrict__ x, const float* __restrict__ g,
                                      const float* __restrict__ b, bf16_t* __restrict__ out) {
  __shared__ float red[64];
  const int row = blockIdx.x, tid = threadIdx.x;     // 64 threads/row
  const float* xr = x + (size_t)row * SE_;
  float s = 0.f, ss = 0.f;
  for (int i = tid; i < SE_; i += 64) { float v = xr[i]; s += v; ss += v * v; }
  red[tid] = s; __syncthreads();
  for (int o = 32; o > 0; o >>= 1) { if (tid < o) red[tid] += red[tid + o]; __syncthreads(); }
  const float mean = red[0] / SE_; __syncthreads();
  red[tid] = ss; __syncthreads();
  for (int o = 32; o > 0; o >>= 1) { if (tid < o) red[tid] += red[tid + o]; __syncthreads(); }
  const float var = red[0] / SE_ - mean * mean;
  const float rs  = rsqrtf(var + 1e-5f);
  bf16_t* o = out + (size_t)row * SE_;
  for (int i = tid; i < SE_; i += 64) o[i] = (bf16_t)((xr[i] - mean) * rs * g[i] + b[i]);
}

__global__ void embed_kernel(const int* __restrict__ tok, const float* __restrict__ obs_emb,
                             const float* __restrict__ act_emb, float* __restrict__ x0) {
  size_t tid = (size_t)blockIdx.x * blockDim.x + threadIdx.x;
  if (tid >= (size_t)BT_ * SE_) return;
  const int e = (int)(tid & 63);
  const size_t bts = tid >> 6;
  const int s = (int)(bts % S_);
  const size_t bt = bts / S_;
  const int t0 = tok[bt * S_ + s];
  x0[bt * SE_ + (size_t)s * E_ + e] =
      (s < 16) ? obs_emb[(size_t)t0 * E_ + e] : act_emb[(size_t)t0 * E_ + e];
}

__global__ void f32_to_bf16_kernel(const float* __restrict__ src, bf16_t* __restrict__ dst, size_t n) {
  size_t tid = (size_t)blockIdx.x * blockDim.x + threadIdx.x;
  if (tid < n) dst[tid] = (bf16_t)src[tid];
}

__global__ void zero_bf16_kernel(bf16_t* __restrict__ dst, size_t n) {
  size_t tid = (size_t)blockIdx.x * blockDim.x + threadIdx.x;
  if (tid < n) dst[tid] = (bf16_t)0.0f;
}

// ---------------------------------------------------------------------------
// Host: input map (setup_inputs dict insertion order, depth-first) + launches.
// ---------------------------------------------------------------------------
static inline size_t alignUp256(size_t x) { return (x + 255) & ~(size_t)255; }
static inline int ceilDiv(size_t a, int b) { return (int)((a + b - 1) / b); }

extern "C" void kernel_launch(void* const* d_in, const int* in_sizes, int n_in,
                              void* d_out, int out_size, void* d_ws, size_t ws_size,
                              hipStream_t stream) {
  (void)in_sizes; (void)n_in; (void)out_size; (void)ws_size;

  // --- input index map -----------------------------------------------------
  // 0 tokens, 1 obs_emb, 2 act_emb,
  // per block (stride 26): gamma, beta, 3x cells{w1,b1,w2,b2,wa,ba,wb,bb}
  // 55.. heads, 67.. masks (2 blocks x 3 cells)
  const int* tokens    = (const int*)d_in[0];
  const float* obs_emb = (const float*)d_in[1];
  const float* act_emb = (const float*)d_in[2];
  auto gammaP = [&](int blk) { return (const float*)d_in[3 + 26 * blk]; };
  auto betaP  = [&](int blk) { return (const float*)d_in[4 + 26 * blk]; };
  auto cellP  = [&](int blk, int c, int f) { return (const float*)d_in[5 + 26 * blk + 8 * c + f]; };
  auto maskP  = [&](int blk, int c) { return (const float*)d_in[67 + 3 * blk + c]; };
  const float* obs_w1 = (const float*)d_in[55];  const float* obs_b1 = (const float*)d_in[56];
  const float* obs_w2 = (const float*)d_in[57];  const float* obs_b2 = (const float*)d_in[58];
  const float* rew_w1 = (const float*)d_in[59];  const float* rew_b1 = (const float*)d_in[60];
  const float* rew_w2 = (const float*)d_in[61];  const float* rew_b2 = (const float*)d_in[62];
  const float* end_w1 = (const float*)d_in[63];  const float* end_b1 = (const float*)d_in[64];
  const float* end_w2 = (const float*)d_in[65];  const float* end_b2 = (const float*)d_in[66];

  const int KD[3] = {1664, 960, 1472};   // din + dh per layer
  const int DH[3] = {576, 384, 1088};

  // --- workspace carve -----------------------------------------------------
  char* wp = (char*)d_ws;
  auto alloc = [&](size_t bytes) { void* p = wp; wp += alignUp256(bytes); return p; };
  float*  x0      = (float*)alloc((size_t)BT_ * SE_ * 4);
  float*  x1      = (float*)alloc((size_t)BT_ * SE_ * 4);
  bf16_t* xn      = (bf16_t*)alloc((size_t)BT_ * SE_ * 2);
  bf16_t* xhBf    = (bf16_t*)alloc((size_t)BT_ * SE_ * 2);
  bf16_t* catAll  = (bf16_t*)alloc((size_t)B_ * (576 + 960 + 1472) * 2);
  bf16_t* h0buf   = catAll;
  bf16_t* cat1    = catAll + (size_t)B_ * 576;
  bf16_t* cat2    = cat1 + (size_t)B_ * 960;
  bf16_t* pcell[2][3];
  for (int blk = 0; blk < 2; ++blk)
    for (int c = 0; c < 3; ++c)
      pcell[blk][c] = (bf16_t*)alloc((size_t)KD[c] * 4 * DH[c] * 2);
  bf16_t* pObs1 = (bf16_t*)alloc((size_t)1088 * 1088 * 2);
  bf16_t* pObs2 = (bf16_t*)alloc((size_t)1088 * 8192 * 2);
  bf16_t* pRew1 = (bf16_t*)alloc((size_t)1088 * 64 * 2);
  bf16_t* pEnd1 = (bf16_t*)alloc((size_t)1088 * 64 * 2);
  bf16_t* obsHidBf = (bf16_t*)alloc((size_t)BT_ * 1088 * 2);
  float*  rewHid   = (float*)alloc((size_t)BT_ * 64 * 4);
  float*  endHid   = (float*)alloc((size_t)BT_ * 64 * 4);

  float* outX   = (float*)d_out;                           // (B,T,SE)
  float* outObs = outX + (size_t)BT_ * SE_;                // (B,T,16,512)
  float* outRew = outObs + (size_t)BT_ * 16 * OBSV_;       // (B,T,3)
  float* outEnd = outRew + (size_t)BT_ * 3;                // (B,T,2)

  // --- pack weights (masked, bf16, per-lane WMMA fragments, contiguous k, gate-interleaved) --
  const int gateField[4] = {0, 2, 4, 6};                   // w1,w2,wa,wb
  for (int blk = 0; blk < 2; ++blk)
    for (int c = 0; c < 3; ++c)
      for (int g = 0; g < 4; ++g)
        pack_w_kernel<<<ceilDiv((size_t)KD[c] * DH[c], 256), 256, 0, stream>>>(
            cellP(blk, c, gateField[g]), g < 2 ? maskP(blk, c) : nullptr,
            pcell[blk][c], KD[c], DH[c], KD[c] / 32, 4, g);
  pack_w_kernel<<<ceilDiv((size_t)1088 * 1088, 256), 256, 0, stream>>>(obs_w1, nullptr, pObs1, 1088, 1088, 34, 1, 0);
  pack_w_kernel<<<ceilDiv((size_t)1088 * 8192, 256), 256, 0, stream>>>(obs_w2, nullptr, pObs2, 1088, 8192, 34, 1, 0);
  pack_w_kernel<<<ceilDiv((size_t)1088 * 64, 256), 256, 0, stream>>>(rew_w1, nullptr, pRew1, 1088, 64, 34, 1, 0);
  pack_w_kernel<<<ceilDiv((size_t)1088 * 64, 256), 256, 0, stream>>>(end_w1, nullptr, pEnd1, 1088, 64, 34, 1, 0);

  // --- embedding -----------------------------------------------------------
  embed_kernel<<<ceilDiv((size_t)BT_ * SE_, 256), 256, 0, stream>>>(tokens, obs_emb, act_emb, x0);

  // --- NCP blocks: LayerNorm + 128-step fused-WMMA CfC recurrence ----------
  const float* xin = x0;
  for (int blk = 0; blk < 2; ++blk) {
    layernorm_bf16_kernel<<<BT_, 64, 0, stream>>>(xin, gammaP(blk), betaP(blk), xn);
    zero_bf16_kernel<<<ceilDiv((size_t)B_ * 3008, 256), 256, 0, stream>>>(catAll, (size_t)B_ * 3008);
    float* xoutBlk = (blk == 0) ? x1 : outX;
    for (int t = 0; t < T_; ++t) {
      // layer 0: A = [ xn[:,t,:] (SE) | h0 (576) ] -> hn0 -> cat1[:,0:576], h0
      cfc_step_kernel<<<18, 128, 0, stream>>>(
          xn + (size_t)t * SE_, (long)T_ * SE_, h0buf, 576, SE_, KD[0] / 32,
          pcell[blk][0], cellP(blk, 0, 1), cellP(blk, 0, 3), cellP(blk, 0, 5), cellP(blk, 0, 7),
          cat1, 960, h0buf, 576, nullptr, 0);
      // layer 1: A = cat1 = [hn0|h1] -> hn1 -> cat2[:,0:384], cat1[:,576:960]
      cfc_step_kernel<<<12, 128, 0, stream>>>(
          cat1, 960, cat1, 960, KD[1], KD[1] / 32,
          pcell[blk][1], cellP(blk, 1, 1), cellP(blk, 1, 3), cellP(blk, 1, 5), cellP(blk, 1, 7),
          cat2, 1472, cat1 + 576, 960, nullptr, 0);
      // layer 2 (motor): A = cat2 = [hn1|h2] -> hn2 -> cat2[:,384:], x_out[:,t,:]
      cfc_step_kernel<<<34, 128, 0, stream>>>(
          cat2, 1472, cat2, 1472, KD[2], KD[2] / 32,
          pcell[blk][2], cellP(blk, 2, 1), cellP(blk, 2, 3), cellP(blk, 2, 5), cellP(blk, 2, 7),
          cat2 + 384, 1472, nullptr, 0, xoutBlk + (size_t)t * SE_, (long)T_ * SE_);
    }
    xin = xoutBlk;
  }

  // --- heads ---------------------------------------------------------------
  f32_to_bf16_kernel<<<ceilDiv((size_t)BT_ * SE_, 256), 256, 0, stream>>>(outX, xhBf, (size_t)BT_ * SE_);
  // head_obs: relu(x@w1+b1) @ w2 + b2 -> (B,T,16,512) flat == (4096, 8192)
  head_gemm_kernel<<<1088, 128, 0, stream>>>(xhBf, SE_, 34, pObs1, 1088 / 64, obs_b1,
                                             nullptr, obsHidBf, 1088, 1);
  head_gemm_kernel<<<8192, 128, 0, stream>>>(obsHidBf, 1088, 34, pObs2, 8192 / 64, obs_b2,
                                             outObs, nullptr, 8192, 0);
  // head_rew / head_end
  head_gemm_kernel<<<64, 128, 0, stream>>>(xhBf, SE_, 34, pRew1, 1, rew_b1, rewHid, nullptr, 64, 1);
  head_small_kernel<<<ceilDiv((size_t)BT_ * 3, 256), 256, 0, stream>>>(rewHid, rew_w2, rew_b2, outRew, 3);
  head_gemm_kernel<<<64, 128, 0, stream>>>(xhBf, SE_, 34, pEnd1, 1, end_b1, endHid, nullptr, 64, 1);
  head_small_kernel<<<ceilDiv((size_t)BT_ * 2, 256), 256, 0, stream>>>(endHid, end_w2, end_b2, outEnd, 2);
}
